// SpectralAttention_10402410791708
// MI455X (gfx1250) — matrix operand order repaired and checked by reading, not compiled, not run on hardware
//
#include <hip/hip_runtime.h>
#include <hip/hip_bf16.h>

// MI455X / gfx1250 chunked linear-attention implementation.
// Precision: f16 operands, f32 WMMA accumulation (v_wmma_f32_16x16x32_f16).
// The O(B*T*P*P) intermediate of the reference is replaced by a P*P running
// state per batch (chunked scan) -> ~100 MB total traffic instead of ~537 GB.
//
// Round-1: __launch_bounds__(512,1) killed the VGPR spills; prefetch added.
// Round-2: async global->LDS copies + scalar (readfirstlane) tile origins.
// Round-3: fix async builtin arg types — compiler revealed the signature:
//          (addrspace(1) v4i* gsrc, addrspace(3) v4i* ldst, imm off, imm cpol)

typedef __attribute__((ext_vector_type(16))) _Float16 v16h;
typedef __attribute__((ext_vector_type(8)))  _Float16 v8h;
typedef __attribute__((ext_vector_type(8)))  float    v8f;
typedef __attribute__((ext_vector_type(4)))  int      v4i;

static constexpr int BB = 4;
static constexpr int TT = 2048;
static constexpr int DD = 1024;
static constexpr int PP = 128;

// workspace layout (offsets in _Float16 elements)
static constexpr size_t OFF_WINT = 0;                       // P x D (W_in^T)
static constexpr size_t OFF_WVT  = OFF_WINT + (size_t)PP * DD;
static constexpr size_t OFF_QFT  = OFF_WVT  + (size_t)PP * DD; // P x P (Qf^T)
static constexpr size_t OFF_KFT  = OFF_QFT  + (size_t)PP * PP;
static constexpr size_t OFF_WOT  = OFF_KFT  + (size_t)PP * PP; // D x P (W_o^T)
static constexpr size_t OFF_Q    = OFF_WOT  + (size_t)DD * PP; // (B*T) x P
static constexpr size_t OFF_KD   = OFF_Q    + (size_t)BB * TT * PP;
static constexpr size_t OFF_V    = OFF_KD   + (size_t)BB * TT * PP;
static constexpr size_t OFF_Y    = OFF_V    + (size_t)BB * TT * PP;

#define WMMA_F16(A_, B_, C_) \
  __builtin_amdgcn_wmma_f32_16x16x32_f16(false, (A_), false, (B_), (short)0, (C_), false, false)

#if defined(__gfx1250__) && __has_builtin(__builtin_amdgcn_global_load_async_to_lds_b128)
#define HAS_ASYNC_COPY 1
#else
#define HAS_ASYNC_COPY 0
#endif

// 16-byte global -> LDS copy. Async (no VGPR round-trip, ASYNCcnt) when the
// gfx1250 builtin is available, plain LDS store otherwise.
__device__ inline void async_copy_b128(_Float16* lds_dst, const _Float16* gsrc) {
#if HAS_ASYNC_COPY
  __builtin_amdgcn_global_load_async_to_lds_b128(
      (__attribute__((address_space(1))) v4i*)const_cast<_Float16*>(gsrc),
      (__attribute__((address_space(3))) v4i*)lds_dst,
      0, 0);
#else
  *(v8h*)lds_dst = *(const v8h*)gsrc;
#endif
}

__device__ inline void async_join() {
#if HAS_ASYNC_COPY
#if __has_builtin(__builtin_amdgcn_s_wait_asynccnt)
  __builtin_amdgcn_s_wait_asynccnt(0);
#else
  asm volatile("s_wait_asynccnt 0" ::: "memory");
#endif
#endif
}

__device__ inline v8f vzero() {
  v8f z = {0.f, 0.f, 0.f, 0.f, 0.f, 0.f, 0.f, 0.f};
  return z;
}

// A fragment: 16x32 f16, row-major tile (row stride lda).
// lane m = lane&15 holds row m; half h = lane>>4 selects K half:
// elems e <-> K = k0 + h*16 + e  (one contiguous 32B load)
__device__ inline v16h load_frag_a(const _Float16* Arow0, int lda, int k0, int lane) {
  const int m = lane & 15, h = (lane >> 4) & 1;
  return *(const v16h*)(Arow0 + (size_t)m * lda + k0 + h * 16);
}

// B fragment: 32x16 f16 where BT is stored N-major (BT[n][k] = B[k][n], row
// stride ldb). lane n = lane&15 holds column n; h = lane>>4 selects K half:
// elems e <-> K = k0 + h*16 + e  (one contiguous 32B load)
__device__ inline v16h load_frag_b(const _Float16* BTrow0, int ldb, int k0, int lane) {
  const int n = lane & 15, h = (lane >> 4) & 1;
  return *(const v16h*)(BTrow0 + (size_t)n * ldb + k0 + h * 16);
}

// 32x32 output block per wave = 2x2 grid of 16x16x32 WMMA tiles, accumulated
// over kLen (multiple of 32). A row-major (M x K), BT N-major (N x K).
__device__ inline void gemm32x32(v8f acc[2][2],
                                 const _Float16* A, int lda, int r0,
                                 const _Float16* BT, int ldb, int c0,
                                 int kLen, int lane) {
  for (int k0 = 0; k0 < kLen; k0 += 32) {
    v16h a0 = load_frag_a(A + (size_t)r0 * lda,        lda, k0, lane);
    v16h a1 = load_frag_a(A + (size_t)(r0 + 16) * lda, lda, k0, lane);
    v16h b0 = load_frag_b(BT + (size_t)c0 * ldb,        ldb, k0, lane);
    v16h b1 = load_frag_b(BT + (size_t)(c0 + 16) * ldb, ldb, k0, lane);
    acc[0][0] = WMMA_F16(a0, b0, acc[0][0]);
    acc[0][1] = WMMA_F16(a0, b1, acc[0][1]);
    acc[1][0] = WMMA_F16(a1, b0, acc[1][0]);
    acc[1][1] = WMMA_F16(a1, b1, acc[1][1]);
  }
}

// ---------------------------------------------------------------------------
// Kernel 0: convert weights to f16, transposed into WMMA "BT" (N-major) form.
// ---------------------------------------------------------------------------
__global__ void wcvt_kernel(const float* __restrict__ W_in, const float* __restrict__ W_v,
                            const float* __restrict__ Qf,   const float* __restrict__ Kf,
                            const float* __restrict__ W_o,  _Float16* __restrict__ ws) {
  const int tid = blockIdx.x * blockDim.x + threadIdx.x;
  const int stride = gridDim.x * blockDim.x;
  _Float16* WinT = ws + OFF_WINT;
  _Float16* WvT  = ws + OFF_WVT;
  _Float16* QfT  = ws + OFF_QFT;
  _Float16* KfT  = ws + OFF_KFT;
  _Float16* WoT  = ws + OFF_WOT;
  for (int i = tid; i < PP * DD; i += stride) {  // (p, d)
    int p = i / DD, d = i % DD;
    WinT[i] = (_Float16)W_in[(size_t)d * PP + p];
    WvT[i]  = (_Float16)W_v [(size_t)d * PP + p];
  }
  for (int i = tid; i < PP * PP; i += stride) {  // (n, k)
    int n = i / PP, k = i % PP;
    QfT[i] = (_Float16)Qf[(size_t)k * PP + n];
    KfT[i] = (_Float16)Kf[(size_t)k * PP + n];
  }
  for (int i = tid; i < DD * PP; i += stride) {  // (n over D, k over P)
    int n = i / PP, k = i % PP;
    WoT[i] = (_Float16)W_o[(size_t)k * DD + n];
  }
}

// ---------------------------------------------------------------------------
// Kernel 1: per 128-row tile of (B*T): XP = x@W_in+b_in, V = x@W_v+b_v,
//           Q = XP@Qf, Kd = (XP@Kf)*decay[t].  512 threads = 16 waves.
// dynamic LDS: 96 KB
// ---------------------------------------------------------------------------
__global__ void __launch_bounds__(512, 1)
qkv_kernel(const float* __restrict__ x,
           const float* __restrict__ b_in, const float* __restrict__ b_v,
           const float* __restrict__ decay, _Float16* __restrict__ ws) {
  extern __shared__ _Float16 sm[];
  const int tid = threadIdx.x;
  const int lane = tid & 31;
  const int w = __builtin_amdgcn_readfirstlane(tid >> 5);   // scalar wave id
  const int wr = (w >> 2) * 32;     // wave row origin (SGPR)
  const int wc = (w & 3) * 32;      // wave col origin (SGPR)
  const int R = blockIdx.x * 128;   // flattened (b*T + t) row base

  _Float16* A_l   = sm;             // 128 x 64  x tile (f16)
  _Float16* Bin_l = sm + 8192;      // 128(N) x 64(K) W_in^T tile
  _Float16* Bv_l  = sm + 16384;     // 128(N) x 64(K) W_v^T tile
  const _Float16* WinT = ws + OFF_WINT;
  const _Float16* WvT  = ws + OFF_WVT;

  v8f accP[2][2], accV[2][2];
#pragma unroll
  for (int t = 0; t < 2; ++t)
#pragma unroll
    for (int u = 0; u < 2; ++u) { accP[t][u] = vzero(); accV[t][u] = vzero(); }

  for (int kb = 0; kb < DD; kb += 64) {
    // weight panels are pure f16 copies -> async global->LDS (16B units)
    for (int i = tid * 8; i < 128 * 64; i += 512 * 8) {
      const int n = i >> 6, k = i & 63;
      async_copy_b128(Bin_l + i, WinT + (size_t)n * DD + kb + k);
      async_copy_b128(Bv_l  + i, WvT  + (size_t)n * DD + kb + k);
    }
    // x panel needs f32->f16 conversion: VGPR path
    for (int i = tid; i < 128 * 64; i += 512) {
      int m = i >> 6, k = i & 63;
      A_l[i] = (_Float16)x[(size_t)(R + m) * DD + kb + k];
    }
    // prefetch next x panel into cache while this panel's WMMAs run
    if (kb + 64 < DD) {
      for (int i = tid; i < 128; i += 512) {   // one lane per row
        __builtin_prefetch(&x[(size_t)(R + i) * DD + kb + 64], 0, 0);
      }
    }
    async_join();
    __syncthreads();
    gemm32x32(accP, A_l, 64, wr, Bin_l, 64, wc, 64, lane);
    gemm32x32(accV, A_l, 64, wr, Bv_l,  64, wc, 64, lane);
    __syncthreads();
  }

  // bias add; store V to workspace (f16)
  _Float16* Vg = ws + OFF_V;
  {
    const int n = lane & 15, h = lane >> 4;
#pragma unroll
    for (int t = 0; t < 2; ++t)
#pragma unroll
      for (int u = 0; u < 2; ++u) {
        const int c0 = wc + u * 16;
        const float bi = b_in[c0 + n];
        const float bv = b_v [c0 + n];
#pragma unroll
        for (int v = 0; v < 8; ++v) {
          accP[t][u][v] += bi;
          const float vv = accV[t][u][v] + bv;
          const int m = wr + t * 16 + v + 8 * h;
          Vg[(size_t)(R + m) * PP + c0 + n] = (_Float16)vv;
        }
      }
  }
  __syncthreads();  // done with stage-1 LDS

  // stage 2: Q = XP@Qf, Kd = (XP@Kf)*decay
  _Float16* XP_l = sm;              // 128 x 128
  _Float16* Qf_l = sm + 16384;      // 128 x 128 (N-major)
  _Float16* Kf_l = sm + 32768;      // 128 x 128 (N-major)
  // filter matrices: pure copies -> async
  for (int i = tid * 8; i < PP * PP; i += 512 * 8) {
    async_copy_b128(Qf_l + i, ws + OFF_QFT + i);
    async_copy_b128(Kf_l + i, ws + OFF_KFT + i);
  }
  {
    const int n = lane & 15, h = lane >> 4;
#pragma unroll
    for (int t = 0; t < 2; ++t)
#pragma unroll
      for (int u = 0; u < 2; ++u)
#pragma unroll
        for (int v = 0; v < 8; ++v) {
          const int m = wr + t * 16 + v + 8 * h;
          XP_l[(size_t)m * PP + wc + u * 16 + n] = (_Float16)accP[t][u][v];
        }
  }
  async_join();
  __syncthreads();

  v8f accQ[2][2], accK[2][2];
#pragma unroll
  for (int t = 0; t < 2; ++t)
#pragma unroll
    for (int u = 0; u < 2; ++u) { accQ[t][u] = vzero(); accK[t][u] = vzero(); }
  gemm32x32(accQ, XP_l, 128, wr, Qf_l, 128, wc, 128, lane);
  gemm32x32(accK, XP_l, 128, wr, Kf_l, 128, wc, 128, lane);

  _Float16* Qg = ws + OFF_Q;
  _Float16* Kg = ws + OFF_KD;
  {
    const int n = lane & 15, h = lane >> 4;
#pragma unroll
    for (int t = 0; t < 2; ++t)
#pragma unroll
      for (int u = 0; u < 2; ++u)
#pragma unroll
        for (int v = 0; v < 8; ++v) {
          const int m = wr + t * 16 + v + 8 * h;
          const int row = R + m;
          const int tglob = row & (TT - 1);   // t index within batch (T=2048)
          const float dec = decay[tglob];
          Qg[(size_t)row * PP + wc + u * 16 + n] = (_Float16)accQ[t][u][v];
          Kg[(size_t)row * PP + wc + u * 16 + n] = (_Float16)(accK[t][u][v] * dec);
        }
  }
}

// ---------------------------------------------------------------------------
// Kernel 2: chunked causal linear-attention scan. One workgroup per batch.
// State H (P x P) lives in per-wave f32 accumulators; its f16 transpose is
// re-exported to LDS each chunk for the Q@H inter-chunk term.
// dynamic LDS: 192 KB (fits 320 KB WGP LDS)
// ---------------------------------------------------------------------------
__global__ void __launch_bounds__(512, 1)
scan_kernel(_Float16* __restrict__ ws) {
  extern __shared__ _Float16 sm[];
  _Float16* Q_l   = sm;              // 128 x 128 (t x i)
  _Float16* V_l   = sm + 16384;      // 128 x 128 (s x i)  == BT for S=Q@V^T
  _Float16* VT_l  = sm + 32768;      // 128 x 128 (i x s)  == A  for H update
  _Float16* KdT_l = sm + 49152;      // 128 x 128 (j x s)  == BT for Y/H
  _Float16* Sm_l  = sm + 65536;      // 128 x 128 masked S (t x s)
  _Float16* HT_l  = sm + 81920;      // 128 x 128 H^T (j x i), f16

  const int tid = threadIdx.x;
  const int lane = tid & 31;
  const int w = __builtin_amdgcn_readfirstlane(tid >> 5);
  const int wr = (w >> 2) * 32;
  const int wc = (w & 3) * 32;
  const int b = blockIdx.x;

  const _Float16* Qg = ws + OFF_Q  + (size_t)b * TT * PP;
  const _Float16* Kg = ws + OFF_KD + (size_t)b * TT * PP;
  const _Float16* Vg = ws + OFF_V  + (size_t)b * TT * PP;
  _Float16*       Yg = ws + OFF_Y  + (size_t)b * TT * PP;

  v8f Hreg[2][2];
#pragma unroll
  for (int t = 0; t < 2; ++t)
#pragma unroll
    for (int u = 0; u < 2; ++u) Hreg[t][u] = vzero();
  for (int i = tid; i < PP * PP; i += 512) HT_l[i] = (_Float16)0.f;
  __syncthreads();

  for (int c = 0; c < TT / 128; ++c) {
    const size_t base = (size_t)c * 128 * PP;
    // Q tile is a straight copy -> async global->LDS
    for (int i = tid * 8; i < 128 * PP; i += 512 * 8) {
      async_copy_b128(Q_l + i, Qg + base + i);
    }
    // V / Kd also feed transposed layouts -> VGPR path with LDS scatter
    for (int i = tid; i < 128 * PP; i += 512) {
      const int s = i >> 7, j = i & 127;
      const _Float16 kd = Kg[base + i];
      const _Float16 vv = Vg[base + i];
      V_l[i] = vv;
      VT_l [(size_t)j * 128 + s] = vv;
      KdT_l[(size_t)j * 128 + s] = kd;
    }
    // prefetch next chunk while current chunk's three GEMM phases execute
    if (c + 1 < TT / 128) {
      const size_t nbase = base + (size_t)128 * PP;
      for (int i = tid; i < 128; i += 512) {     // one cacheline probe per row
        __builtin_prefetch(&Qg[nbase + (size_t)i * PP], 0, 0);
        __builtin_prefetch(&Kg[nbase + (size_t)i * PP], 0, 0);
        __builtin_prefetch(&Vg[nbase + (size_t)i * PP], 0, 0);
      }
    }
    async_join();
    __syncthreads();

    // Phase S: S[t,s] = Q_t . V_s , causal-masked, f16 into Sm_l.
    // Tiles entirely above the diagonal are skipped (scalar branch: wr/wc are
    // SGPRs, so this is an s_cbranch and EXEC stays all-ones for the WMMAs).
    {
      const int n = lane & 15, h = lane >> 4;
#pragma unroll
      for (int t2 = 0; t2 < 2; ++t2)
#pragma unroll
        for (int u = 0; u < 2; ++u) {
          const int r0 = wr + t2 * 16, c0 = wc + u * 16;
          v8f a = vzero();
          if (c0 <= r0 + 15) {
            for (int k0 = 0; k0 < 128; k0 += 32) {
              v16h af = load_frag_a(Q_l + (size_t)r0 * 128, 128, k0, lane);
              v16h bf = load_frag_b(V_l + (size_t)c0 * 128, 128, k0, lane);
              a = WMMA_F16(af, bf, a);
            }
          }
#pragma unroll
          for (int v = 0; v < 8; ++v) {
            const int trow = r0 + v + 8 * h;
            const int scol = c0 + n;
            const float val = (scol <= trow) ? a[v] : 0.f;
            Sm_l[(size_t)trow * 128 + scol] = (_Float16)val;
          }
        }
    }
    __syncthreads();

    // Phase Y: Y = Q @ H_prev (BT = HT_l) + Sm @ Kd (BT = KdT_l)
    {
      v8f acc[2][2];
#pragma unroll
      for (int t2 = 0; t2 < 2; ++t2)
#pragma unroll
        for (int u = 0; u < 2; ++u) acc[t2][u] = vzero();
      gemm32x32(acc, Q_l,  128, wr, HT_l,  128, wc, 128, lane);
      gemm32x32(acc, Sm_l, 128, wr, KdT_l, 128, wc, 128, lane);
      const int n = lane & 15, h = lane >> 4;
#pragma unroll
      for (int t2 = 0; t2 < 2; ++t2)
#pragma unroll
        for (int u = 0; u < 2; ++u)
#pragma unroll
          for (int v = 0; v < 8; ++v) {
            const int m = wr + t2 * 16 + v + 8 * h;
            Yg[base + (size_t)m * PP + wc + u * 16 + n] = (_Float16)acc[t2][u][v];
          }
    }
    __syncthreads();  // everyone done reading HT_l before it is rewritten

    // Phase H: Hreg += V^T @ Kd ; export f16 H^T for next chunk
    gemm32x32(Hreg, VT_l, 128, wr, KdT_l, 128, wc, 128, lane);
    {
      const int n = lane & 15, h = lane >> 4;
#pragma unroll
      for (int t2 = 0; t2 < 2; ++t2)
#pragma unroll
        for (int u = 0; u < 2; ++u)
#pragma unroll
          for (int v = 0; v < 8; ++v) {
            const int irow = wr + t2 * 16 + v + 8 * h;   // i
            const int jcol = wc + u * 16 + n;            // j
            HT_l[(size_t)jcol * 128 + irow] = (_Float16)Hreg[t2][u][v];
          }
    }
    __syncthreads();  // HT complete; chunk buffers free for next load
  }
}

// ---------------------------------------------------------------------------
// Kernel 3: out = Y @ W_o + b_o  (f32 output).  dynamic LDS: 64 KB
// ---------------------------------------------------------------------------
__global__ void __launch_bounds__(512, 1)
outp_kernel(const float* __restrict__ b_o,
            const _Float16* __restrict__ ws,
            float* __restrict__ out) {
  extern __shared__ _Float16 sm[];
  _Float16* Y_l = sm;           // 128 rows x 128 (K=P)
  _Float16* W_l = sm + 16384;   // 128 cols x 128 (K=P), N-major (W_o^T tile)
  const int tid = threadIdx.x;
  const int lane = tid & 31;
  const int w = __builtin_amdgcn_readfirstlane(tid >> 5);
  const int wr = (w >> 2) * 32;
  const int wc = (w & 3) * 32;
  const int R  = blockIdx.y * 128;   // row tile in (B*T)
  const int C0 = blockIdx.x * 128;   // col tile in D

  const _Float16* Yg  = ws + OFF_Y;
  const _Float16* WoT = ws + OFF_WOT;
  // both tiles are pure copies -> async global->LDS
  for (int i = tid * 8; i < 128 * PP; i += 512 * 8) {
    async_copy_b128(Y_l + i, Yg  + (size_t)R  * PP + i);
    async_copy_b128(W_l + i, WoT + (size_t)C0 * PP + i);
  }
  async_join();
  __syncthreads();

  v8f acc[2][2];
#pragma unroll
  for (int t = 0; t < 2; ++t)
#pragma unroll
    for (int u = 0; u < 2; ++u) acc[t][u] = vzero();
  gemm32x32(acc, Y_l, 128, wr, W_l, 128, wc, 128, lane);

  const int n = lane & 15, h = lane >> 4;
#pragma unroll
  for (int t2 = 0; t2 < 2; ++t2)
#pragma unroll
    for (int u = 0; u < 2; ++u) {
      const float bo = b_o[C0 + wc + u * 16 + n];
#pragma unroll
      for (int v = 0; v < 8; ++v) {
        const int m = wr + t2 * 16 + v + 8 * h;
        out[(size_t)(R + m) * DD + C0 + wc + u * 16 + n] = acc[t2][u][v] + bo;
      }
    }
}

// ---------------------------------------------------------------------------
extern "C" void kernel_launch(void* const* d_in, const int* in_sizes, int n_in,
                              void* d_out, int out_size, void* d_ws, size_t ws_size,
                              hipStream_t stream) {
  const float* x     = (const float*)d_in[0];
  const float* Qf    = (const float*)d_in[1];
  const float* Kf    = (const float*)d_in[2];
  const float* W_in  = (const float*)d_in[3];
  const float* b_in  = (const float*)d_in[4];
  const float* W_v   = (const float*)d_in[5];
  const float* b_v   = (const float*)d_in[6];
  const float* W_o   = (const float*)d_in[7];
  const float* b_o   = (const float*)d_in[8];
  const float* decay = (const float*)d_in[9];
  _Float16* ws = (_Float16*)d_ws;
  float* out = (float*)d_out;

  wcvt_kernel<<<512, 256, 0, stream>>>(W_in, W_v, Qf, Kf, W_o, ws);
  qkv_kernel<<<(BB * TT) / 128, 512, 96 * 1024, stream>>>(x, b_in, b_v, decay, ws);
  // NOTE: 192 KB dynamic LDS — within gfx1250's 320 KB/WGP; a runtime build
  // would set hipFuncAttributeMaxDynamicSharedMemorySize before launch.
  scan_kernel<<<BB, 512, 192 * 1024, stream>>>(ws);
  dim3 gout(DD / 128, (BB * TT) / 128);
  outp_kernel<<<gout, 512, 64 * 1024, stream>>>(b_o, ws, out);
}
